// MultiHeadedAttention_18983755448471
// MI455X (gfx1250) — matrix operand order repaired and checked by reading, not compiled, not run on hardware
//
#include <hip/hip_runtime.h>

#define D_MODEL 1024
#define SEQ     2048
#define BATCH   4
#define NH      16
#define DKH     64
#define LN_EPS  1e-5f

typedef __attribute__((ext_vector_type(16))) __bf16 v16bf;
typedef __attribute__((ext_vector_type(8)))  __bf16 v8bf;
typedef __attribute__((ext_vector_type(4)))  __bf16 v4bf;
typedef __attribute__((ext_vector_type(8)))  float  v8f;
typedef __attribute__((ext_vector_type(4)))  float  v4f;

// ---------------------------------------------------------------------------
// WMMA fragment loaders (wave32, layouts per CDNA5 ISA §7.12.2)
// ---------------------------------------------------------------------------

// 16-bit A matrix 16x32 (MxK), row-major source with row stride `ld` elements.
// lane<16:  elems 0..7 = K 0..7,  elems 8..15 = K 16..23  (M = lane)
// lane>=16: elems 0..7 = K 8..15, elems 8..15 = K 24..31  (M = lane-16)
static __device__ inline v16bf load_a_frag(const __bf16* __restrict__ p, int ld, int lane) {
  const __bf16* q = p + (size_t)(lane & 15) * ld + ((lane >> 4) * 8);
  v8bf lo = *(const v8bf*)(q);
  v8bf hi = *(const v8bf*)(q + 16);
  v16bf a;
#pragma unroll
  for (int j = 0; j < 8; ++j) { a[j] = lo[j]; a[j + 8] = hi[j]; }
  return a;
}

// 16-bit B matrix 32x16 (KxN) where B(k,n) = M[n*ld + k] (source holds N rows of
// contiguous K). frag elem j<8: K = 8*(lane>>4) + j; j>=8: K = 16 + 8*(lane>>4) + (j-8); N = lane&15.
static __device__ inline v16bf load_bT_frag(const __bf16* __restrict__ p, int ld, int lane) {
  const __bf16* q = p + (size_t)(lane & 15) * ld + ((lane >> 4) * 8);
  v8bf lo = *(const v8bf*)(q);
  v8bf hi = *(const v8bf*)(q + 16);
  v16bf b;
#pragma unroll
  for (int j = 0; j < 8; ++j) { b[j] = lo[j]; b[j + 8] = hi[j]; }
  return b;
}

static __device__ inline v8f wmma_bf16(v16bf a, v16bf b, v8f c) {
  // (neg_a, A, neg_b, B, c_mod, C, reuse_a, reuse_b)
  return __builtin_amdgcn_wmma_f32_16x16x32_bf16(false, a, false, b, (short)0, c,
                                                 false, false);
}

// ---------------------------------------------------------------------------
// f32 -> bf16 conversion, 4 elements per thread (all sizes are multiples of 4)
// ---------------------------------------------------------------------------
__global__ void f32_to_bf16_kernel(const float* __restrict__ in,
                                   __bf16* __restrict__ out, int nVec4) {
  int i = blockIdx.x * blockDim.x + threadIdx.x;
  if (i < nVec4) {
    v4f v = *(const v4f*)(in + (size_t)4 * i);
    v4bf o;
#pragma unroll
    for (int j = 0; j < 4; ++j) o[j] = (__bf16)v[j];
    *(v4bf*)(out + (size_t)4 * i) = o;
  }
}

// ---------------------------------------------------------------------------
// Q/K projection: Y = X * W^T, scattered into [B,H,S,DKH] bf16 head layout.
// One wave computes one 16x16 tile; 4 waves per block over N tiles.
// ---------------------------------------------------------------------------
__global__ __launch_bounds__(128) void proj_qk_kernel(
    const __bf16* __restrict__ X,     // [B*S, D]
    const __bf16* __restrict__ W,     // [D, D] row-major (torch Linear)
    __bf16* __restrict__ OutHead) {   // [B, H, S, DKH]
  const int lane = threadIdx.x & 31;
  const int wave = threadIdx.x >> 5;
  const int row0 = blockIdx.x * 16;
  const int col0 = (blockIdx.y * 4 + wave) * 16;
  const __bf16* xp = X + (size_t)row0 * D_MODEL;
  const __bf16* wp = W + (size_t)col0 * D_MODEL;
  v8f c = {};
  for (int k0 = 0; k0 < D_MODEL; k0 += 32) {
    v16bf a  = load_a_frag(xp + k0, D_MODEL, lane);
    v16bf bf = load_bT_frag(wp + k0, D_MODEL, lane);
    c = wmma_bf16(a, bf, c);
  }
  const int nn = lane & 15, hiw = lane >> 4;
  const int e = col0 + nn;
  const int h = e >> 6;
  const int dcol = e & 63;
#pragma unroll
  for (int r = 0; r < 8; ++r) {
    const int sflat = row0 + r + 8 * hiw;   // flat b*S + s
    const int bb = sflat >> 11;             // S = 2048
    const int s  = sflat & (SEQ - 1);
    OutHead[(((size_t)(bb * NH + h)) * SEQ + s) * DKH + dcol] = (__bf16)c[r];
  }
}

// ---------------------------------------------------------------------------
// V projection: same GEMM but writes V TRANSPOSED per head: [B,H,DKH,S].
// This makes the P*V B-fragments contiguous 16B loads in the flash kernel.
// ---------------------------------------------------------------------------
__global__ __launch_bounds__(128) void proj_v_kernel(
    const __bf16* __restrict__ X, const __bf16* __restrict__ W,
    __bf16* __restrict__ OutT) {      // [B, H, DKH, S]
  const int lane = threadIdx.x & 31;
  const int wave = threadIdx.x >> 5;
  const int row0 = blockIdx.x * 16;
  const int col0 = (blockIdx.y * 4 + wave) * 16;
  const __bf16* xp = X + (size_t)row0 * D_MODEL;
  const __bf16* wp = W + (size_t)col0 * D_MODEL;
  v8f c = {};
  for (int k0 = 0; k0 < D_MODEL; k0 += 32) {
    v16bf a  = load_a_frag(xp + k0, D_MODEL, lane);
    v16bf bf = load_bT_frag(wp + k0, D_MODEL, lane);
    c = wmma_bf16(a, bf, c);
  }
  const int nn = lane & 15, hiw = lane >> 4;
  const int e = col0 + nn;
  const int h = e >> 6;
  const int dcol = e & 63;
#pragma unroll
  for (int r = 0; r < 8; ++r) {
    const int sflat = row0 + r + 8 * hiw;
    const int bb = sflat >> 11;
    const int s  = sflat & (SEQ - 1);
    OutT[(((size_t)(bb * NH + h)) * DKH + dcol) * SEQ + s] = (__bf16)c[r];
  }
}

// ---------------------------------------------------------------------------
// Flash attention: one wave per (b, h, 32-row q tile). Two 16-row M-subtiles
// share all K/V fragment loads (2x arithmetic intensity). Online softmax,
// P transposed C-layout -> A-layout through a per-wave LDS staging buffer.
// ---------------------------------------------------------------------------
__global__ __launch_bounds__(32) void flash_attn_kernel(
    const __bf16* __restrict__ Qh,   // [B,H,S,DKH]
    const __bf16* __restrict__ Kh,   // [B,H,S,DKH]
    const __bf16* __restrict__ Vt,   // [B,H,DKH,S]
    const int* __restrict__ mask,    // [B,S,S]
    __bf16* __restrict__ Ctx) {      // [B,S,D] head-concatenated
  __shared__ __bf16 Pbuf[32 * 32];
  const int lane = threadIdx.x & 31;
  const int qt = blockIdx.x, h = blockIdx.y, b = blockIdx.z;
  const int q0 = qt * 32;
  const size_t headOff = ((size_t)(b * NH + h)) * SEQ * DKH;
  const __bf16* Qb = Qh + headOff;
  const __bf16* Kb = Kh + headOff;
  const __bf16* Vb = Vt + headOff;   // [DKH, SEQ]
  const int nn = lane & 15;
  const int hiw = lane >> 4;

  // Q fragments: 2 M-subtiles x 2 K-chunks
  v16bf aq[2][2];
#pragma unroll
  for (int t = 0; t < 2; ++t) {
    aq[t][0] = load_a_frag(Qb + (size_t)(q0 + 16 * t) * DKH,      DKH, lane);
    aq[t][1] = load_a_frag(Qb + (size_t)(q0 + 16 * t) * DKH + 32, DKH, lane);
  }

  float m[2][8], l[2][8];
  v8f acc[2][4];
#pragma unroll
  for (int t = 0; t < 2; ++t) {
#pragma unroll
    for (int r = 0; r < 8; ++r) { m[t][r] = -__builtin_inff(); l[t][r] = 0.f; }
#pragma unroll
    for (int n = 0; n < 4; ++n) acc[t][n] = {};
  }

  for (int k0 = 0; k0 < SEQ; k0 += 32) {
    // K fragments for columns [k0, k0+32): shared by both M-subtiles
    v16bf bkA0 = load_bT_frag(Kb + (size_t)k0 * DKH,             DKH, lane);
    v16bf bkA1 = load_bT_frag(Kb + (size_t)k0 * DKH + 32,        DKH, lane);
    v16bf bkB0 = load_bT_frag(Kb + (size_t)(k0 + 16) * DKH,      DKH, lane);
    v16bf bkB1 = load_bT_frag(Kb + (size_t)(k0 + 16) * DKH + 32, DKH, lane);

    v8f s[2][2];
#pragma unroll
    for (int t = 0; t < 2; ++t) {
      v8f s0 = {}, s1 = {};
      s0 = wmma_bf16(aq[t][0], bkA0, s0);
      s0 = wmma_bf16(aq[t][1], bkA1, s0);
      s1 = wmma_bf16(aq[t][0], bkB0, s1);
      s1 = wmma_bf16(aq[t][1], bkB1, s1);
      s[t][0] = s0; s[t][1] = s1;
    }

#pragma unroll
    for (int t = 0; t < 2; ++t) {
      const int* mbase = mask + ((size_t)b * SEQ + q0 + 16 * t) * SEQ;
#pragma unroll
      for (int r = 0; r < 8; ++r) {
        const int qrow = r + 8 * hiw;                 // row within subtile
        const int* mr = mbase + (size_t)qrow * SEQ + k0 + nn;
        float v0 = (mr[0]  == 1) ? -1e9f : s[t][0][r] * 0.125f;  // 1/sqrt(64)
        float v1 = (mr[16] == 1) ? -1e9f : s[t][1][r] * 0.125f;
        float mt = fmaxf(v0, v1);
#pragma unroll
        for (int d = 1; d < 16; d <<= 1) mt = fmaxf(mt, __shfl_xor(mt, d, 32));
        const float mn = fmaxf(m[t][r], mt);
        const float alpha = __expf(m[t][r] - mn);
        m[t][r] = mn;
        const float e0 = __expf(v0 - mn);
        const float e1 = __expf(v1 - mn);
        float lt = e0 + e1;
#pragma unroll
        for (int d = 1; d < 16; d <<= 1) lt += __shfl_xor(lt, d, 32);
        l[t][r] = l[t][r] * alpha + lt;
#pragma unroll
        for (int n = 0; n < 4; ++n) acc[t][n][r] *= alpha;
        // stage P (C layout) into LDS as bf16, row-major 32x32
        Pbuf[(t * 16 + qrow) * 32 + nn]      = (__bf16)e0;
        Pbuf[(t * 16 + qrow) * 32 + 16 + nn] = (__bf16)e1;
      }
    }
    __syncthreads();                                 // single wave -> S_NOP
    v16bf ap0 = load_a_frag(Pbuf,           32, lane);  // subtile 0, A layout
    v16bf ap1 = load_a_frag(Pbuf + 16 * 32, 32, lane);  // subtile 1
    // V fragments from V^T: contiguous 16B loads, shared by both subtiles
    v16bf bv0 = load_bT_frag(Vb + (size_t) 0 * SEQ + k0, SEQ, lane);
    v16bf bv1 = load_bT_frag(Vb + (size_t)16 * SEQ + k0, SEQ, lane);
    v16bf bv2 = load_bT_frag(Vb + (size_t)32 * SEQ + k0, SEQ, lane);
    v16bf bv3 = load_bT_frag(Vb + (size_t)48 * SEQ + k0, SEQ, lane);
    acc[0][0] = wmma_bf16(ap0, bv0, acc[0][0]);
    acc[0][1] = wmma_bf16(ap0, bv1, acc[0][1]);
    acc[0][2] = wmma_bf16(ap0, bv2, acc[0][2]);
    acc[0][3] = wmma_bf16(ap0, bv3, acc[0][3]);
    acc[1][0] = wmma_bf16(ap1, bv0, acc[1][0]);
    acc[1][1] = wmma_bf16(ap1, bv1, acc[1][1]);
    acc[1][2] = wmma_bf16(ap1, bv2, acc[1][2]);
    acc[1][3] = wmma_bf16(ap1, bv3, acc[1][3]);
    __syncthreads();
  }

#pragma unroll
  for (int t = 0; t < 2; ++t) {
#pragma unroll
    for (int r = 0; r < 8; ++r) {
      const int qrow = q0 + 16 * t + r + 8 * hiw;
      const float inv = 1.0f / l[t][r];
      __bf16* op = Ctx + ((size_t)b * SEQ + qrow) * D_MODEL + h * DKH + nn;
      op[0]  = (__bf16)(acc[t][0][r] * inv);
      op[16] = (__bf16)(acc[t][1][r] * inv);
      op[32] = (__bf16)(acc[t][2][r] * inv);
      op[48] = (__bf16)(acc[t][3][r] * inv);
    }
  }
}

// ---------------------------------------------------------------------------
// Output projection + residual: O = Ctx * Wo^T + residual (f32 out)
// ---------------------------------------------------------------------------
__global__ __launch_bounds__(128) void proj_o_kernel(
    const __bf16* __restrict__ Ctx, const __bf16* __restrict__ Wo,
    const float* __restrict__ resid, float* __restrict__ Of) {
  const int lane = threadIdx.x & 31;
  const int wave = threadIdx.x >> 5;
  const int row0 = blockIdx.x * 16;
  const int col0 = (blockIdx.y * 4 + wave) * 16;
  const __bf16* xp = Ctx + (size_t)row0 * D_MODEL;
  const __bf16* wp = Wo + (size_t)col0 * D_MODEL;
  v8f c = {};
  for (int k0 = 0; k0 < D_MODEL; k0 += 32) {
    v16bf a  = load_a_frag(xp + k0, D_MODEL, lane);
    v16bf bf = load_bT_frag(wp + k0, D_MODEL, lane);
    c = wmma_bf16(a, bf, c);
  }
  const int nn = lane & 15, hiw = lane >> 4;
#pragma unroll
  for (int r = 0; r < 8; ++r) {
    const size_t idx = (size_t)(row0 + r + 8 * hiw) * D_MODEL + col0 + nn;
    Of[idx] = c[r] + resid[idx];
  }
}

// ---------------------------------------------------------------------------
// Row LayerNorm: one block (256 threads) per row of 1024 f32 elements.
// ---------------------------------------------------------------------------
__global__ __launch_bounds__(256) void layernorm_kernel(
    const float* __restrict__ X, const float* __restrict__ gamma,
    const float* __restrict__ beta, float* __restrict__ out) {
  const int row = blockIdx.x;
  const float* x = X + (size_t)row * D_MODEL;
  float s = 0.f, s2 = 0.f;
  for (int i = threadIdx.x; i < D_MODEL; i += 256) {
    const float v = x[i];
    s += v; s2 += v * v;
  }
#pragma unroll
  for (int d = 1; d < 32; d <<= 1) {
    s  += __shfl_xor(s, d, 32);
    s2 += __shfl_xor(s2, d, 32);
  }
  __shared__ float ps[8], ps2[8];
  const int wv = threadIdx.x >> 5;
  if ((threadIdx.x & 31) == 0) { ps[wv] = s; ps2[wv] = s2; }
  __syncthreads();
  if (threadIdx.x == 0) {
    float a = 0.f, b2 = 0.f;
#pragma unroll
    for (int i = 0; i < 8; ++i) { a += ps[i]; b2 += ps2[i]; }
    ps[0] = a; ps2[0] = b2;
  }
  __syncthreads();
  const float mu  = ps[0] * (1.0f / D_MODEL);
  const float var = ps2[0] * (1.0f / D_MODEL) - mu * mu;
  const float inv = rsqrtf(var + LN_EPS);
  for (int i = threadIdx.x; i < D_MODEL; i += 256) {
    out[(size_t)row * D_MODEL + i] = (x[i] - mu) * inv * gamma[i] + beta[i];
  }
}

// ---------------------------------------------------------------------------
// Host launcher. Workspace usage (<= 109,051,904 bytes):
//   Xq/Xk/Xv bf16 [B*S*D]   : 3 * 16 MiB
//   Wq/Wk/Wv/Wo bf16 [D*D]  : 4 * 2  MiB
//   Qh/Kh bf16 [B,H,S,dk], Vt bf16 [B,H,dk,S] : 3 * 16 MiB
//   Ctx bf16 reuses Xq; O f32 reuses Xk+Xv.
// ---------------------------------------------------------------------------
extern "C" void kernel_launch(void* const* d_in, const int* in_sizes, int n_in,
                              void* d_out, int out_size, void* d_ws, size_t ws_size,
                              hipStream_t stream) {
  (void)in_sizes; (void)n_in; (void)out_size; (void)ws_size;
  const float* query = (const float*)d_in[0];
  const float* key_i = (const float*)d_in[1];
  const float* value = (const float*)d_in[2];
  const int*   mask  = (const int*)d_in[3];
  const float* W_Q   = (const float*)d_in[4];
  const float* W_K   = (const float*)d_in[5];
  const float* W_V   = (const float*)d_in[6];
  const float* W_O   = (const float*)d_in[7];
  const float* gamma = (const float*)d_in[8];
  const float* beta  = (const float*)d_in[9];

  const size_t BSD = (size_t)BATCH * SEQ * D_MODEL;  // 8,388,608
  const size_t DD  = (size_t)D_MODEL * D_MODEL;      // 1,048,576

  __bf16* Xq = (__bf16*)d_ws;
  __bf16* Xk = Xq + BSD;
  __bf16* Xv = Xk + BSD;
  __bf16* Wq = Xv + BSD;
  __bf16* Wk = Wq + DD;
  __bf16* Wv = Wk + DD;
  __bf16* Wo = Wv + DD;
  __bf16* Qh = Wo + DD;
  __bf16* Kh = Qh + BSD;
  __bf16* Vt = Kh + BSD;
  __bf16* Ctx  = Xq;           // reuse: Xq dead after projections
  float*  Of32 = (float*)Xk;   // reuse: Xk..Xv region == BSD f32 exactly

  const int CT = 256;
  f32_to_bf16_kernel<<<(int)((BSD / 4 + CT - 1) / CT), CT, 0, stream>>>(query, Xq, (int)(BSD / 4));
  f32_to_bf16_kernel<<<(int)((BSD / 4 + CT - 1) / CT), CT, 0, stream>>>(key_i, Xk, (int)(BSD / 4));
  f32_to_bf16_kernel<<<(int)((BSD / 4 + CT - 1) / CT), CT, 0, stream>>>(value, Xv, (int)(BSD / 4));
  f32_to_bf16_kernel<<<(int)((DD / 4 + CT - 1) / CT), CT, 0, stream>>>(W_Q, Wq, (int)(DD / 4));
  f32_to_bf16_kernel<<<(int)((DD / 4 + CT - 1) / CT), CT, 0, stream>>>(W_K, Wk, (int)(DD / 4));
  f32_to_bf16_kernel<<<(int)((DD / 4 + CT - 1) / CT), CT, 0, stream>>>(W_V, Wv, (int)(DD / 4));
  f32_to_bf16_kernel<<<(int)((DD / 4 + CT - 1) / CT), CT, 0, stream>>>(W_O, Wo, (int)(DD / 4));

  dim3 gProj(BATCH * SEQ / 16, D_MODEL / 64);
  proj_qk_kernel<<<gProj, 128, 0, stream>>>(Xq, Wq, Qh);
  proj_qk_kernel<<<gProj, 128, 0, stream>>>(Xk, Wk, Kh);
  proj_v_kernel<<<gProj, 128, 0, stream>>>(Xv, Wv, Vt);

  flash_attn_kernel<<<dim3(SEQ / 32, NH, BATCH), 32, 0, stream>>>(Qh, Kh, Vt, mask, Ctx);

  proj_o_kernel<<<gProj, 128, 0, stream>>>(Ctx, Wo, query, Of32);

  layernorm_kernel<<<BATCH * SEQ, 256, 0, stream>>>(Of32, gamma, beta, (float*)d_out);
}
